// Dual_SelfAttention_69406671503576
// MI455X (gfx1250) — matrix operand order repaired and checked by reading, not compiled, not run on hardware
//
#include <hip/hip_runtime.h>
#include <hip/hip_bf16.h>

// CDNA5 (gfx1250, wave32) implementation of the dual self-attention block.
//
// Exact algebraic simplification: E_c and E_u are identical computations in the
// reference (source bug), so the gate A = sigmoid(0) = 0.5 exactly and the
// W_d / h_vec / ReLU branch is dead:  out = 0.5*out1 + 0.5*out2 + inputs.
//
// Pipeline (all GEMMs on v_wmma_f32_16x16x32_f16, f32 accumulate):
//   1. prep:    xh  = f16(inputs+pe), xuh = f16(inputs2+pe)   [padded L=1008]
//   2. wtconv:  W^T -> f16 [k][n] row-major (Wq,Wk,Wv,Wqu,fc_w)
//   3. gemm16:  qh/vh/quh (f16 row-major), khT (f16 transposed [b][d][key])
//   4. attn x2: per (batch, 16-query tile): S = q @ k^T in LDS, causal-masked
//               softmax in LDS (in-place), out += 0.5 * P @ V
//   5. ln:      t = o + inputs; LayerNorm; yh = f16(t)
//   6. gemm16:  d_out = yh @ fc_w^T + fc_b (f32)
//
// Register-blocking: every WMMA site computes a 16x64 strip per wave (4 output
// tiles share one A fragment) to raise WMMA density per VMEM instruction.
//
// Workspace: ~293 MB laid out inside d_ws.

typedef _Float16 v16h __attribute__((ext_vector_type(16)));
typedef float    v8f  __attribute__((ext_vector_type(8)));
typedef _Float16 f16x2 __attribute__((ext_vector_type(2)));

#define DD 1024
#define LL 1000
#define NBATCH 16
#define LPAD 1008          // 63 tiles of 16
#define QTILES 63

static __device__ __forceinline__ v8f wmma_f16(const v16h& a, const v16h& b, v8f c) {
  // (neg_a, A, neg_b, B, c_mod, C, reuse_a, reuse_b)
  return __builtin_amdgcn_wmma_f32_16x16x32_f16(false, a, false, b, (short)0, c, false, false);
}

// K offset inside a 16x32 f16 A fragment for register-pair v (0..7), before the
// lane-half (+8) adjustment. Pairs are contiguous in K.
static __device__ __forceinline__ int a_koff(int v) {
  return ((v & 4) ? 16 : 0) + (v & 3) * 2;
}

static __device__ __forceinline__ v16h load_afrag(const _Float16* ap) {
  v16h af;
#pragma unroll
  for (int v = 0; v < 8; ++v) {
    f16x2 p = *(const f16x2*)(ap + a_koff(v));
    af[2 * v] = p[0]; af[2 * v + 1] = p[1];
  }
  return af;
}

static __device__ __forceinline__ v16h load_bfrag(const _Float16* bp, int ld) {
  v16h bf;
#pragma unroll
  for (int v = 0; v < 8; ++v) {
    bf[2 * v] = bp[(size_t)a_koff(v) * ld]; bf[2 * v + 1] = bp[(size_t)(a_koff(v) + 1) * ld];
  }
  return bf;
}

// ---------------------------------------------------------------------------
// 1. prep: xh = f16(inputs + pe), xuh = f16(inputs2 + pe), zero pad rows.
// pe[pos][d] = (d even ? sin : cos)(pos * 10000^(-d/1024))
// ---------------------------------------------------------------------------
__global__ __launch_bounds__(256) void prep_kernel(
    const float* __restrict__ x1, const float* __restrict__ x2,
    _Float16* __restrict__ xh, _Float16* __restrict__ xuh) {
  size_t idx = (size_t)blockIdx.x * 256 + threadIdx.x;
  if (idx >= (size_t)NBATCH * LPAD * DD) return;
  int d = (int)(idx & (DD - 1));
  int r = (int)(idx >> 10);
  int b = r / LPAD, pos = r % LPAD;
  _Float16 a = (_Float16)0.0f, c = (_Float16)0.0f;
  if (pos < LL) {
    float rate = __expf(-(float)d * (9.210340371976184f / 1024.0f)); // ln(1e4)/1024
    float ang  = (float)pos * rate;
    float pe   = (d & 1) ? __cosf(ang) : __sinf(ang);
    size_t si  = ((size_t)(b * LL + pos)) * DD + d;
    a = (_Float16)(x1[si] + pe);
    c = (_Float16)(x2[si] + pe);
  }
  xh[idx]  = a;
  xuh[idx] = c;
}

// ---------------------------------------------------------------------------
// 2. weight transpose + f16 convert: Wt[k][n] = (f16) W[n][k]  (1024x1024)
// ---------------------------------------------------------------------------
__global__ __launch_bounds__(256) void wtconv_kernel(
    const float* __restrict__ W, _Float16* __restrict__ Wt) {
  int idx = blockIdx.x * 256 + threadIdx.x;      // < 1024*1024
  int k = idx >> 10, n = idx & (DD - 1);
  Wt[idx] = (_Float16)W[n * DD + k];
}

// ---------------------------------------------------------------------------
// 3/6. Generic GEMM: C[M,1024] = A[M,1024](f16) * Bt (Bt[k][n] f16).
// One wave per 16x64 output strip (4 tiles, shared A fragment), K=1024 in 32
// steps of wmma 16x16x32.
// mode 0: f16 row-major out     mode 1: f16 transposed-per-batch (for khT)
// mode 2: f32 out + bias (final FC)
// ---------------------------------------------------------------------------
__global__ __launch_bounds__(256) void gemm16_kernel(
    const _Float16* __restrict__ A, const _Float16* __restrict__ Bt,
    _Float16* __restrict__ outH, _Float16* __restrict__ outT,
    float* __restrict__ outF, const float* __restrict__ bias,
    int Mtiles, int mode) {
  const int wv   = threadIdx.x >> 5;
  const int lane = threadIdx.x & 31;
  const int grp  = blockIdx.x * 8 + wv;          // 16x64 strip index
  if (grp >= Mtiles * 16) return;
  const int mt    = grp >> 4;
  const int nbase = (grp & 15) * 64;
  const int m16 = lane & 15, hsel = lane >> 4;

  const _Float16* arow = A + ((size_t)(mt * 16 + m16)) * DD + hsel * 8;
  v8f acc[4] = {};
  for (int kc = 0; kc < 32; ++kc) {
    v16h af = load_afrag(arow + kc * 32);
    const _Float16* brow = Bt + ((size_t)(kc * 32 + hsel * 8)) * DD + nbase + m16;
#pragma unroll
    for (int j = 0; j < 4; ++j) {
      v16h bf = load_bfrag(brow + j * 16, DD);
      acc[j] = wmma_f16(af, bf, acc[j]);
    }
  }
#pragma unroll
  for (int j = 0; j < 4; ++j) {
#pragma unroll
    for (int r = 0; r < 8; ++r) {
      int m  = r + 8 * hsel;
      int gm = mt * 16 + m;
      int n  = nbase + j * 16 + m16;
      if (mode == 0) {
        outH[(size_t)gm * DD + n] = (_Float16)acc[j][r];
      } else if (mode == 1) {
        int bb = gm / LPAD, key = gm % LPAD;
        outT[((size_t)(bb * DD + n)) * LPAD + key] = (_Float16)acc[j][r];
      } else {
        outF[(size_t)gm * DD + n] = acc[j][r] + bias[n];
      }
    }
  }
}

// ---------------------------------------------------------------------------
// 4. Causal attention for one (batch, 16-query tile).  256 threads = 8 waves.
//    Stage A: S[16][keys] in LDS via wmma, Q fragment reused across 4 K-tiles.
//    Stage B: row softmax (max/sum via 16-lane shuffles), P written in place.
//    Stage C: out strip (16 x 64 per wave-group) = P @ V, P fragment reused.
//    accumulate=0: O = 0.5*PV ; accumulate=1: O += 0.5*PV.
// ---------------------------------------------------------------------------
__global__ __launch_bounds__(256) void attn_kernel(
    const _Float16* __restrict__ Q, const _Float16* __restrict__ KT,
    const _Float16* __restrict__ V, float* __restrict__ O, int accumulate) {
  __shared__ _Float16 SP[16][1024];   // logits, then probabilities (in place)

  const int qt  = blockIdx.x % QTILES;
  const int bat = blockIdx.x / QTILES;
  const int wv = threadIdx.x >> 5, lane = threadIdx.x & 31;
  const int m16 = lane & 15, hsel = lane >> 4;

  // ---- Stage A: scores; wave handles groups of 4 key tiles ----
  const _Float16* qrow = Q + ((size_t)(bat * LPAD + qt * 16 + m16)) * DD + hsel * 8;
  const int ktiles = qt + 1;
  for (int g = wv; g * 4 < ktiles; g += 8) {
    const int kt0 = g * 4;
    v8f acc[4] = {};
    for (int kc = 0; kc < 32; ++kc) {
      v16h af = load_afrag(qrow + kc * 32);
      const _Float16* brow = KT + ((size_t)(bat * DD + kc * 32 + hsel * 8)) * LPAD + kt0 * 16 + m16;
#pragma unroll
      for (int j = 0; j < 4; ++j) {
        if (kt0 + j < ktiles) {                 // wave-uniform guard
          v16h bf = load_bfrag(brow + j * 16, LPAD);
          acc[j] = wmma_f16(af, bf, acc[j]);
        }
      }
    }
#pragma unroll
    for (int j = 0; j < 4; ++j) {
      int kt = kt0 + j;
      if (kt < ktiles) {
#pragma unroll
        for (int r = 0; r < 8; ++r) {
          int m = r + 8 * hsel;
          int key = kt * 16 + m16;
          int query = qt * 16 + m;
          SP[m][kt * 16 + m16] = (_Float16)((key <= query) ? acc[j][r] : -65504.0f);
        }
      }
    }
  }
  __syncthreads();

  const int Kcause = (qt + 1) * 16;
  const int chunks = (Kcause + 31) >> 5;
  const int Kcover = chunks << 5;

  // ---- Stage B: softmax per row (16 threads per row, same half-wave) ----
  {
    const int row = threadIdx.x >> 4, c = threadIdx.x & 15;
    float mx = -1e30f;
    for (int key = c; key < Kcover; key += 16) {
      float s = (key < Kcause) ? (float)SP[row][key] : -1e30f;
      mx = fmaxf(mx, s);
    }
    for (int off = 1; off < 16; off <<= 1) mx = fmaxf(mx, __shfl_xor(mx, off, 16));
    float sum = 0.0f;
    for (int key = c; key < Kcover; key += 16)
      if (key < Kcause) sum += __expf((float)SP[row][key] - mx);
    for (int off = 1; off < 16; off <<= 1) sum += __shfl_xor(sum, off, 16);
    float inv = 1.0f / sum;
    for (int key = c; key < Kcover; key += 16) {
      float p = (key < Kcause) ? __expf((float)SP[row][key] - mx) * inv : 0.0f;
      SP[row][key] = (_Float16)p;   // in place; this thread owns these slots
    }
  }
  __syncthreads();

  // ---- Stage C: O(16 x 1024) = P @ V, groups of 4 d-tiles per wave ----
  for (int g = wv; g < 16; g += 8) {
    const int dt0 = g * 4;
    v8f acc[4] = {};
    for (int ch = 0; ch < chunks; ++ch) {
      v16h af = load_afrag(&SP[m16][ch * 32 + hsel * 8]);
      const _Float16* brow = V + ((size_t)(bat * LPAD + ch * 32 + hsel * 8)) * DD + dt0 * 16 + m16;
#pragma unroll
      for (int j = 0; j < 4; ++j) {
        v16h bf = load_bfrag(brow + j * 16, DD);
        acc[j] = wmma_f16(af, bf, acc[j]);
      }
    }
#pragma unroll
    for (int j = 0; j < 4; ++j) {
#pragma unroll
      for (int r = 0; r < 8; ++r) {
        int m = r + 8 * hsel;
        size_t oi = ((size_t)(bat * LPAD + qt * 16 + m)) * DD + (dt0 + j) * 16 + m16;
        float val = 0.5f * acc[j][r];
        if (accumulate) val += O[oi];
        O[oi] = val;
      }
    }
  }
}

// ---------------------------------------------------------------------------
// 5. t = O + inputs; LayerNorm over D; yh = f16 result.  One 256-block/row.
// ---------------------------------------------------------------------------
__global__ __launch_bounds__(256) void ln_kernel(
    const float* __restrict__ O, const float* __restrict__ X,
    const float* __restrict__ g, const float* __restrict__ be,
    _Float16* __restrict__ Y) {
  int r = blockIdx.x;                 // 0 .. 15999
  int b = r / LL, pos = r % LL;
  const float* op = O + ((size_t)(b * LPAD + pos)) * DD;
  const float* xp = X + (size_t)r * DD;
  float t[4], s1 = 0.0f, s2 = 0.0f;
#pragma unroll
  for (int k = 0; k < 4; ++k) {
    int d = threadIdx.x + k * 256;
    float v = op[d] + xp[d];
    t[k] = v; s1 += v; s2 += v * v;
  }
  for (int off = 16; off >= 1; off >>= 1) {
    s1 += __shfl_xor(s1, off, 32);
    s2 += __shfl_xor(s2, off, 32);
  }
  __shared__ float r1[8], r2[8];
  if ((threadIdx.x & 31) == 0) { r1[threadIdx.x >> 5] = s1; r2[threadIdx.x >> 5] = s2; }
  __syncthreads();
  float tot1 = 0.0f, tot2 = 0.0f;
#pragma unroll
  for (int i = 0; i < 8; ++i) { tot1 += r1[i]; tot2 += r2[i]; }
  float mean = tot1 * (1.0f / 1024.0f);
  float var  = tot2 * (1.0f / 1024.0f) - mean * mean;
  float rs   = rsqrtf(var + 1e-5f);
  _Float16* yp = Y + (size_t)r * DD;
#pragma unroll
  for (int k = 0; k < 4; ++k) {
    int d = threadIdx.x + k * 256;
    yp[d] = (_Float16)((t[k] - mean) * rs * g[d] + be[d]);
  }
}

// ---------------------------------------------------------------------------
extern "C" void kernel_launch(void* const* d_in, const int* in_sizes, int n_in,
                              void* d_out, int out_size, void* d_ws, size_t ws_size,
                              hipStream_t stream) {
  (void)in_sizes; (void)n_in; (void)out_size; (void)ws_size;

  const float* inputs  = (const float*)d_in[0];
  const float* inputs2 = (const float*)d_in[1];
  const float* Wq      = (const float*)d_in[2];
  const float* Wqu     = (const float*)d_in[3];
  const float* Wk      = (const float*)d_in[4];
  const float* Wv      = (const float*)d_in[5];
  // d_in[6] = W_d, d_in[7] = h_vec: dead (gate A == 0.5 exactly, see header)
  const float* fcw     = (const float*)d_in[8];
  const float* fcb     = (const float*)d_in[9];
  const float* lng     = (const float*)d_in[10];
  const float* lnb     = (const float*)d_in[11];
  float* out = (float*)d_out;

  char* ws = (char*)d_ws;
  const size_t HB = (size_t)NBATCH * LPAD * DD * 2;   // f16 activation buffer
  const size_t WB = (size_t)DD * DD * 2;              // f16 weight buffer
  _Float16* xh   = (_Float16*)(ws + 0 * HB);
  _Float16* xuh  = (_Float16*)(ws + 1 * HB);
  _Float16* qh   = (_Float16*)(ws + 2 * HB);
  _Float16* khT  = (_Float16*)(ws + 3 * HB);          // [b][d][key]
  _Float16* vh   = (_Float16*)(ws + 4 * HB);
  _Float16* quh  = (_Float16*)(ws + 5 * HB);
  _Float16* wqT  = (_Float16*)(ws + 6 * HB + 0 * WB);
  _Float16* wkT  = (_Float16*)(ws + 6 * HB + 1 * WB);
  _Float16* wvT  = (_Float16*)(ws + 6 * HB + 2 * WB);
  _Float16* wquT = (_Float16*)(ws + 6 * HB + 3 * WB);
  _Float16* fcT  = (_Float16*)(ws + 6 * HB + 4 * WB);
  float*    o    = (float*)   (ws + 6 * HB + 5 * WB); // [b][LPAD][D] f32
  _Float16* yh   = (_Float16*)(ws + 6 * HB + 5 * WB + (size_t)NBATCH * LPAD * DD * 4);

  // 1. inputs + positional encoding -> f16 (padded)
  prep_kernel<<<64512, 256, 0, stream>>>(inputs, inputs2, xh, xuh);

  // 2. weights -> f16 transposed [k][n]
  wtconv_kernel<<<4096, 256, 0, stream>>>(Wq,  wqT);
  wtconv_kernel<<<4096, 256, 0, stream>>>(Wk,  wkT);
  wtconv_kernel<<<4096, 256, 0, stream>>>(Wv,  wvT);
  wtconv_kernel<<<4096, 256, 0, stream>>>(Wqu, wquT);
  wtconv_kernel<<<4096, 256, 0, stream>>>(fcw, fcT);

  // 3. projections (16128 rows -> 1008 M-tiles -> 16128 strips -> 2016 blocks)
  gemm16_kernel<<<2016, 256, 0, stream>>>(xh,  wqT,  qh,  nullptr, nullptr, nullptr, 1008, 0);
  gemm16_kernel<<<2016, 256, 0, stream>>>(xh,  wvT,  vh,  nullptr, nullptr, nullptr, 1008, 0);
  gemm16_kernel<<<2016, 256, 0, stream>>>(xuh, wquT, quh, nullptr, nullptr, nullptr, 1008, 0);
  gemm16_kernel<<<2016, 256, 0, stream>>>(xh,  wkT,  nullptr, khT, nullptr, nullptr, 1008, 1);

  // 4. two attention passes into the shared 0.5-weighted accumulator
  attn_kernel<<<NBATCH * QTILES, 256, 0, stream>>>(qh,  khT, vh, o, 0);
  attn_kernel<<<NBATCH * QTILES, 256, 0, stream>>>(quh, khT, vh, o, 1);

  // 5. residual + layernorm -> f16
  ln_kernel<<<NBATCH * LL, 256, 0, stream>>>(o, inputs, lng, lnb, yh);

  // 6. final FC with bias -> d_out (16000 rows -> 1000 M-tiles -> 2000 blocks)
  gemm16_kernel<<<2000, 256, 0, stream>>>(yh, fcT, nullptr, nullptr, out, fcb, 1000, 2);
}
